// ParallelLRU_2233382994292
// MI455X (gfx1250) — matrix operand order repaired and checked by reading, not compiled, not run on hardware
//
#include <hip/hip_runtime.h>
#include <hip/hip_bf16.h>
#include <stdint.h>

// ---------------- problem constants (from reference) ----------------
#define D_DIM   1024
#define T_DIM   8192
#define B_DIM   8
#define NT      (B_DIM * T_DIM)     // 65536 token rows
#define CHUNK   128
#define NCHUNK  (T_DIM / CHUNK)     // 64
#define BLK_M   32                  // rows per block (2 WMMA row-tiles)
#define LN_EPSF 1e-5f

typedef __bf16 bf16_t;
typedef __attribute__((ext_vector_type(16))) __bf16 v16bf;
typedef __attribute__((ext_vector_type(8)))  float  v8f;

// round-to-nearest-even f32 -> bf16
__device__ __forceinline__ bf16_t f2bf(float f) {
  union { float f; uint32_t u; } v; v.f = f;
  uint32_t r = v.u + 0x7FFFu + ((v.u >> 16) & 1u);
  uint16_t h = (uint16_t)(r >> 16);
  bf16_t b;
  __builtin_memcpy(&b, &h, 2);
  return b;
}

// ---------------- prep: lam vectors + weight conversion ----------------
__global__ __launch_bounds__(256) void prep_vec_kernel(
    const float* __restrict__ llr, float* __restrict__ lam,
    float* __restrict__ oml, float* __restrict__ lamL) {
  int d = blockIdx.x * 256 + threadIdx.x;
  if (d >= D_DIM) return;
  float s = 1.0f / (1.0f + expf(-llr[d]));
  lam[d] = s;
  oml[d] = 1.0f - s;
  float p = s;                       // lam^128 via 7 squarings
  #pragma unroll
  for (int i = 0; i < 7; ++i) p = p * p;
  lamL[d] = p;
}

__global__ __launch_bounds__(256) void convert_w_kernel(
    const float* __restrict__ src, bf16_t* __restrict__ dst) {
  int i = blockIdx.x * 256 + threadIdx.x;   // grid covers D*D exactly
  dst[i] = f2bf(src[i]);
}

// ---------------- GEMM core: 32 rows x 1024 cols per block ----------------
// 8 waves; wave w owns cols [w*128, w*128+128) as 8 WMMA n-tiles.
// Two A row-tiles (rows 0-15, 16-31) share every B fragment: 2x L2 reuse.
__device__ __forceinline__ void gemm_rowtile32_accum(
    const bf16_t* __restrict__ Asm,   // LDS, 32 x D row-major bf16
    const bf16_t* __restrict__ W,     // global bf16, [D][D], out-dim major
    int lane, int ncol0, v8f acc0[8], v8f acc1[8]) {
  const int half = lane >> 4;      // 0: k 0-7 and 16-23, 1: k 8-15 and 24-31
  const int l16  = lane & 15;
  for (int k0 = 0; k0 < D_DIM; k0 += 32) {
    // A fragments per ISA 16-bit A layout: VGPR i holds a K pair
    v16bf a0, a1;
    #pragma unroll
    for (int i = 0; i < 8; ++i) {
      int kb = k0 + ((i < 4) ? 2 * i : 16 + 2 * (i - 4)) + half * 8;
      a0[2 * i]     = Asm[l16 * D_DIM + kb];
      a0[2 * i + 1] = Asm[l16 * D_DIM + kb + 1];
      a1[2 * i]     = Asm[(16 + l16) * D_DIM + kb];
      a1[2 * i + 1] = Asm[(16 + l16) * D_DIM + kb + 1];
    }
    #pragma unroll
    for (int j = 0; j < 8; ++j) {
      int n = ncol0 + j * 16 + l16;            // output column = W row
      const bf16_t* wrow = W + (size_t)n * D_DIM;
      v16bf bfr;
      #pragma unroll
      for (int i = 0; i < 8; ++i) {
        int kb = k0 + ((i < 4) ? 2 * i : 16 + 2 * (i - 4)) + half * 8;  // B: same K order as A
        bfr[2 * i]     = wrow[kb];
        bfr[2 * i + 1] = wrow[kb + 1];
      }
      acc0[j] = __builtin_amdgcn_wmma_f32_16x16x32_bf16(
          false, a0, false, bfr, (short)0, acc0[j], false, false);
      acc1[j] = __builtin_amdgcn_wmma_f32_16x16x32_bf16(
          false, a1, false, bfr, (short)0, acc1[j], false, false);
    }
  }
}

// ---------------- GEMM-1: values = (1-lam) * (x @ Wb^T + bb) ----------------
__global__ __launch_bounds__(256) void gemm_b_kernel(
    const float* __restrict__ x, const bf16_t* __restrict__ Wb_bf,
    const float* __restrict__ bb, const float* __restrict__ oml,
    float* __restrict__ values) {
  __shared__ bf16_t Asm[BLK_M * D_DIM];          // 64 KB
  const int row0 = blockIdx.x * BLK_M;
  const int tid  = threadIdx.x;
  for (int i = tid; i < BLK_M * D_DIM; i += 256) {
    int r = i >> 10, c = i & (D_DIM - 1);
    Asm[i] = f2bf(x[(size_t)(row0 + r) * D_DIM + c]);
  }
  __syncthreads();

  const int wave = tid >> 5, lane = tid & 31;
  const int half = lane >> 4, l16 = lane & 15;
  const int ncol0 = wave * 128;
  v8f acc0[8], acc1[8];
  #pragma unroll
  for (int j = 0; j < 8; ++j) { acc0[j] = (v8f)(0.0f); acc1[j] = (v8f)(0.0f); }

  gemm_rowtile32_accum(Asm, Wb_bf, lane, ncol0, acc0, acc1);

  #pragma unroll
  for (int j = 0; j < 8; ++j) {
    int c = ncol0 + j * 16 + l16;
    float g = oml[c], bias = bb[c];
    #pragma unroll
    for (int i = 0; i < 8; ++i) {
      int r = row0 + i + 8 * half;               // C layout: M = i + 8*half
      values[(size_t)r * D_DIM + c]        = g * (acc0[j][i] + bias);
      values[(size_t)(r + 16) * D_DIM + c] = g * (acc1[j][i] + bias);
    }
  }
}

// ---------------- chunked parallel scan h[t] = lam*h[t-1] + v[t] ----------------
__global__ __launch_bounds__(256) void scan_local_kernel(
    const float* __restrict__ values, const float* __restrict__ lam,
    float* __restrict__ local_end) {
  int idx = blockIdx.x * 256 + threadIdx.x;
  int d = idx & (D_DIM - 1);
  int ch = (idx >> 10) & (NCHUNK - 1);
  int b = idx >> 16;
  float l = lam[d];
  const float* vp = values + ((size_t)b * T_DIM + (size_t)ch * CHUNK) * D_DIM + d;
  float h = 0.0f;
  for (int t = 0; t < CHUNK; ++t) h = l * h + vp[(size_t)t * D_DIM];
  local_end[idx] = h;
}

__global__ __launch_bounds__(256) void scan_carry_kernel(
    const float* __restrict__ local_end, const float* __restrict__ lamL,
    float* __restrict__ carry_in) {
  int idx = blockIdx.x * 256 + threadIdx.x;    // b*D + d
  int d = idx & (D_DIM - 1);
  int b = idx >> 10;
  float lL = lamL[d];
  float c = 0.0f;
  for (int ch = 0; ch < NCHUNK; ++ch) {
    size_t o = ((size_t)b * NCHUNK + ch) * D_DIM + d;
    carry_in[o] = c;                 // h at end of previous chunk
    c = lL * c + local_end[o];
  }
}

__global__ __launch_bounds__(256) void scan_apply_kernel(
    const float* __restrict__ values, const float* __restrict__ lam,
    const float* __restrict__ carry_in, bf16_t* __restrict__ h_bf) {
  int idx = blockIdx.x * 256 + threadIdx.x;
  int d = idx & (D_DIM - 1);
  int ch = (idx >> 10) & (NCHUNK - 1);
  int b = idx >> 16;
  float l = lam[d];
  size_t base = ((size_t)b * T_DIM + (size_t)ch * CHUNK) * D_DIM + d;
  float h = carry_in[idx];
  for (int t = 0; t < CHUNK; ++t) {
    h = l * h + values[base + (size_t)t * D_DIM];
    h_bf[base + (size_t)t * D_DIM] = f2bf(h);
  }
}

// ---------------- GEMM-2 + residual + fused LayerNorm ----------------
// A (bf16) staged via GLOBAL_LOAD_ASYNC_TO_LDS (ASYNCcnt path); Z overlaps A
// in the same 128 KB LDS buffer (A dead after MAC loop, barrier-separated).
__global__ __launch_bounds__(256) void gemm_c_ln_kernel(
    const bf16_t* __restrict__ h_bf, const bf16_t* __restrict__ Wc_bf,
    const float* __restrict__ bc, const float* __restrict__ x,
    const float* __restrict__ gamma, const float* __restrict__ beta,
    float* __restrict__ out) {
  __shared__ __align__(16) char smem[BLK_M * D_DIM * sizeof(float)]; // 128 KB
  bf16_t* Asm = (bf16_t*)smem;          // first 64 KB during MAC phase
  float*  Zsm = (float*)smem;           // full 128 KB during LN phase
  __shared__ float redS[BLK_M][9];
  __shared__ float redQ[BLK_M][9];
  __shared__ float muS[BLK_M], rstdS[BLK_M];

  const int row0 = blockIdx.x * BLK_M;
  const int tid  = threadIdx.x;

  // async stage A: 32 rows x 1024 bf16 = 64 KB = 256 thr x 16 x b128
  {
    const bf16_t* gbase = h_bf + (size_t)row0 * D_DIM;
    #pragma unroll
    for (int it = 0; it < 16; ++it) {
      int e = (it * 256 + tid) * 8;                 // 8 bf16 per b128
      unsigned lds_off = (unsigned)(uintptr_t)(&Asm[e]);
      const bf16_t* gp = gbase + e;
      asm volatile("global_load_async_to_lds_b128 %0, %1, off"
                   :: "v"(lds_off), "v"(gp) : "memory");
    }
    asm volatile("s_wait_asynccnt 0x0" ::: "memory");
  }
  __syncthreads();

  const int wave = tid >> 5, lane = tid & 31;
  const int half = lane >> 4, l16 = lane & 15;
  const int ncol0 = wave * 128;
  v8f acc0[8], acc1[8];
  #pragma unroll
  for (int j = 0; j < 8; ++j) { acc0[j] = (v8f)(0.0f); acc1[j] = (v8f)(0.0f); }

  gemm_rowtile32_accum(Asm, Wc_bf, lane, ncol0, acc0, acc1);

  __syncthreads();   // all waves done reading A before Z overwrites it

  // z = out + bc + x  -> LDS for full-row LayerNorm
  #pragma unroll
  for (int j = 0; j < 8; ++j) {
    int c = ncol0 + j * 16 + l16;
    float bias = bc[c];
    #pragma unroll
    for (int i = 0; i < 8; ++i) {
      int r = i + 8 * half;
      Zsm[r * D_DIM + c] =
          acc0[j][i] + bias + x[(size_t)(row0 + r) * D_DIM + c];
      Zsm[(r + 16) * D_DIM + c] =
          acc1[j][i] + bias + x[(size_t)(row0 + r + 16) * D_DIM + c];
    }
  }
  __syncthreads();

  // per-row mean/var: 8 threads per row (32 rows), each sums 128 cols
  {
    int rr = tid >> 3, sub = tid & 7;
    float s = 0.0f, q = 0.0f;
    for (int c = sub; c < D_DIM; c += 8) {
      float v = Zsm[rr * D_DIM + c];
      s += v; q += v * v;
    }
    redS[rr][sub] = s; redQ[rr][sub] = q;
  }
  __syncthreads();
  if ((tid & 7) == 0) {
    int rr = tid >> 3;
    float s = 0.0f, q = 0.0f;
    #pragma unroll
    for (int i = 0; i < 8; ++i) { s += redS[rr][i]; q += redQ[rr][i]; }
    float mu = s * (1.0f / D_DIM);
    float var = q * (1.0f / D_DIM) - mu * mu;
    muS[rr] = mu;
    rstdS[rr] = rsqrtf(var + LN_EPSF);
  }
  __syncthreads();

  // normalize + affine, coalesced store
  for (int i = tid; i < BLK_M * D_DIM; i += 256) {
    int r = i >> 10, c = i & (D_DIM - 1);
    float z = (Zsm[i] - muS[r]) * rstdS[r] * gamma[c] + beta[c];
    out[(size_t)(row0 + r) * D_DIM + c] = z;
  }
}

// ---------------- launcher ----------------
extern "C" void kernel_launch(void* const* d_in, const int* in_sizes, int n_in,
                              void* d_out, int out_size, void* d_ws, size_t ws_size,
                              hipStream_t stream) {
  (void)in_sizes; (void)n_in; (void)out_size; (void)ws_size;
  const float* x     = (const float*)d_in[0];
  const float* llr   = (const float*)d_in[1];
  const float* Wb    = (const float*)d_in[2];
  const float* bb    = (const float*)d_in[3];
  const float* Wc    = (const float*)d_in[4];
  const float* bc    = (const float*)d_in[5];
  const float* gamma = (const float*)d_in[6];
  const float* beta  = (const float*)d_in[7];
  float* out = (float*)d_out;

  char* ws = (char*)d_ws;
  bf16_t* Wb_bf = (bf16_t*)ws;  ws += (size_t)D_DIM * D_DIM * 2;   // 2 MB
  bf16_t* Wc_bf = (bf16_t*)ws;  ws += (size_t)D_DIM * D_DIM * 2;   // 2 MB
  float* lam    = (float*)ws;   ws += (size_t)D_DIM * 4;
  float* oml    = (float*)ws;   ws += (size_t)D_DIM * 4;
  float* lamL   = (float*)ws;   ws += (size_t)D_DIM * 4;
  float* local_end = (float*)ws; ws += (size_t)B_DIM * NCHUNK * D_DIM * 4; // 2 MB
  float* carry_in  = (float*)ws; ws += (size_t)B_DIM * NCHUNK * D_DIM * 4; // 2 MB
  bf16_t* h_bf  = (bf16_t*)ws;  ws += (size_t)NT * D_DIM * 2;      // 128 MB
  float* values = out;  // reuse output buffer as pre-scan scratch (overwritten by final kernel)

  prep_vec_kernel<<<(D_DIM + 255) / 256, 256, 0, stream>>>(llr, lam, oml, lamL);
  convert_w_kernel<<<(D_DIM * D_DIM) / 256, 256, 0, stream>>>(Wb, Wb_bf);
  convert_w_kernel<<<(D_DIM * D_DIM) / 256, 256, 0, stream>>>(Wc, Wc_bf);

  gemm_b_kernel<<<NT / BLK_M, 256, 0, stream>>>(x, Wb_bf, bb, oml, values);

  scan_local_kernel<<<(B_DIM * NCHUNK * D_DIM) / 256, 256, 0, stream>>>(values, lam, local_end);
  scan_carry_kernel<<<(B_DIM * D_DIM) / 256, 256, 0, stream>>>(local_end, lamL, carry_in);
  scan_apply_kernel<<<(B_DIM * NCHUNK * D_DIM) / 256, 256, 0, stream>>>(values, lam, carry_in, h_bf);

  gemm_c_ln_kernel<<<NT / BLK_M, 256, 0, stream>>>(h_bf, Wc_bf, bc, x, gamma, beta, out);
}
